// Ac4kAttentionOp_50379966382838
// MI455X (gfx1250) — 8-bit matrix operand order repaired and checked by reading, compiled, not run on hardware
//
#include <hip/hip_runtime.h>
#include <hip/hip_bf16.h>


#define BH_   16
#define N_    2048
#define D_    128
#define KT_   128
#define QT_   128
#define NT_   (N_ / KT_)

#define SM_K    0
#define SM_VT   32768
#define SM_P    65536
#define SM_VS   81920
#define SMEM_BYTES 82944

typedef __attribute__((ext_vector_type(8)))  int   v8i;
typedef __attribute__((ext_vector_type(16))) int   v16i;
typedef __attribute__((ext_vector_type(8)))  float v8f;

__device__ __forceinline__ unsigned int f32_to_fp8(float x) {
  return ((unsigned int)__builtin_amdgcn_cvt_pk_fp8_f32(x, 0.0f, 0, false)) & 0xFFu;
}

__device__ __forceinline__ float hmax16(float v) {
#pragma unroll
  for (int m = 1; m < 16; m <<= 1) v = fmaxf(v, __shfl_xor(v, m, 32));
  return v;
}
__device__ __forceinline__ float hsum16(float v) {
#pragma unroll
  for (int m = 1; m < 16; m <<= 1) v += __shfl_xor(v, m, 32);
  return v;
}

__device__ __forceinline__ void async_ld_b128(const void* g, unsigned lds_off) {
  asm volatile("global_load_async_to_lds_b128 %0, %1, off"
               :: "v"(lds_off), "v"((unsigned long long)(uintptr_t)g)
               : "memory");
}
__device__ __forceinline__ void async_ld_b32(const void* g, unsigned lds_off) {
  asm volatile("global_load_async_to_lds_b32 %0, %1, off"
               :: "v"(lds_off), "v"((unsigned long long)(uintptr_t)g)
               : "memory");
}
__device__ __forceinline__ void wait_async_le9() {
  asm volatile("s_wait_asynccnt 0x9" ::: "memory");
}
__device__ __forceinline__ void wait_async_le0() {
  asm volatile("s_wait_asynccnt 0x0" ::: "memory");
}


__global__ void kmean_kernel(const float* __restrict__ k, float* __restrict__ kmean) {
  const int bh = blockIdx.x, d = threadIdx.x;
  float s = 0.f;
  for (int n = 0; n < N_; ++n) s += k[((size_t)bh * N_ + n) * D_ + d];
  kmean[bh * D_ + d] = s * (1.0f / N_);
}

__global__ void quantk_kernel(const float* __restrict__ k, const float* __restrict__ kmean,
                              signed char* __restrict__ k8, float* __restrict__ kscale) {
  __shared__ float red[256];
  const int tid = threadIdx.x;
  const int bh = blockIdx.x >> 4, kb = blockIdx.x & 15;
  const size_t base = ((size_t)bh * N_ + (size_t)kb * 128) * D_;
  float amax = 0.f;
  for (int i = 0; i < 64; ++i) {
    const int idx = tid + i * 256;
    const int d = idx & (D_ - 1);
    const float v = k[base + idx] - kmean[bh * D_ + d];
    amax = fmaxf(amax, fabsf(v));
  }
  red[tid] = amax; __syncthreads();
  for (int s = 128; s > 0; s >>= 1) {
    if (tid < s) red[tid] = fmaxf(red[tid], red[tid + s]);
    __syncthreads();
  }
  const float scale = red[0] * (1.0f / 127.0f);
  if (tid == 0) kscale[bh * 16 + kb] = scale;
  const float inv = (scale == 0.f) ? 1.0f : 1.0f / scale;
  for (int i = 0; i < 64; ++i) {
    const int idx = tid + i * 256;
    const int d = idx & (D_ - 1);
    const float v = k[base + idx] - kmean[bh * D_ + d];
    float y = truncf(v * inv);
    y = fminf(fmaxf(y, -128.f), 127.f);
    k8[base + idx] = (signed char)(int)y;
  }
}

__global__ void quantq_kernel(const float* __restrict__ q,
                              signed char* __restrict__ q8, float* __restrict__ qscale) {
  __shared__ float red[256];
  const int tid = threadIdx.x;
  const int bh = blockIdx.x >> 5, qb = blockIdx.x & 31;
  const size_t base = ((size_t)bh * N_ + (size_t)qb * 64) * D_;
  float amax = 0.f;
  for (int i = 0; i < 32; ++i) amax = fmaxf(amax, fabsf(q[base + tid + i * 256]));
  red[tid] = amax; __syncthreads();
  for (int s = 128; s > 0; s >>= 1) {
    if (tid < s) red[tid] = fmaxf(red[tid], red[tid + s]);
    __syncthreads();
  }
  const float scale = red[0] * (1.0f / 127.0f);
  if (tid == 0) qscale[bh * 32 + qb] = scale;
  const float inv = (scale == 0.f) ? 1.0f : 1.0f / scale;
  for (int i = 0; i < 32; ++i) {
    const int idx = tid + i * 256;
    float y = truncf(q[base + idx] * inv);
    y = fminf(fmaxf(y, -128.f), 127.f);
    q8[base + idx] = (signed char)(int)y;
  }
}

__global__ void quantv_kernel(const float* __restrict__ v,
                              signed char* __restrict__ v8t, float* __restrict__ vscale) {
  const int tid = threadIdx.x;
  const int wave = tid >> 5, lane = tid & 31;
  const int tok = blockIdx.x * 8 + wave;
  const int bh = tok >> 11, n = tok & (N_ - 1);
  const size_t vbase = ((size_t)bh * N_ + n) * D_;
  float x[4];
  float amax = 0.f;
#pragma unroll
  for (int i = 0; i < 4; ++i) {
    x[i] = v[vbase + lane + 32 * i];
    amax = fmaxf(amax, fabsf(x[i]));
  }
#pragma unroll
  for (int m = 1; m < 32; m <<= 1) amax = fmaxf(amax, __shfl_xor(amax, m, 32));
  const float sf = amax * (1.0f / 448.0f);
  if (lane == 0) vscale[tok] = sf;
  const float inv = (sf == 0.f) ? 1.0f : 1.0f / sf;
#pragma unroll
  for (int i = 0; i < 4; ++i) {
    const int d = lane + 32 * i;
    v8t[(size_t)bh * D_ * N_ + (size_t)d * N_ + n] = (signed char)f32_to_fp8(x[i] * inv);
  }
}


__device__ __forceinline__ void stage_tile_async(
    const signed char* __restrict__ k8, const signed char* __restrict__ v8t,
    const float* __restrict__ vscale, char* smem, int bh, int kt, int buf, int tid) {
  const size_t kbase = ((size_t)bh * N_ + (size_t)kt * KT_) * D_;
  const unsigned ldsk = (unsigned)(uintptr_t)(smem + SM_K + buf * 16384 + tid * 16);
#pragma unroll
  for (int i = 0; i < 4; ++i)
    async_ld_b128(k8 + kbase + tid * 16 + i * 4096, ldsk + i * 4096);

  const int vrow = tid >> 1, voff = (tid & 1) * 64;
  const size_t vbase = (size_t)bh * D_ * N_ + (size_t)vrow * N_ + (size_t)kt * KT_ + voff;
  const unsigned ldsv =
      (unsigned)(uintptr_t)(smem + SM_VT + buf * 16384 + vrow * KT_ + voff);
#pragma unroll
  for (int i = 0; i < 4; ++i)
    async_ld_b128(v8t + vbase + i * 16, ldsv + i * 16);

  const int vsid = tid & 127;
  async_ld_b32(vscale + bh * N_ + kt * KT_ + vsid,
               (unsigned)(uintptr_t)(smem + SM_VS + buf * 512 + vsid * 4));
}

__global__ __launch_bounds__(256, 1) void attn_fwd_kernel(
    const signed char* __restrict__ q8, const signed char* __restrict__ k8,
    const signed char* __restrict__ v8t, const float* __restrict__ qscale,
    const float* __restrict__ kscale, const float* __restrict__ vscale,
    __hip_bfloat16* __restrict__ out) {
  extern __shared__ __align__(16) char smem[];

  const int tid = threadIdx.x;
  const int wave = tid >> 5, lane = tid & 31;
  const int hi = lane >> 4, ln = lane & 15;
  const int qt = blockIdx.x, bh = blockIdx.y;
  const float sm_scale = 0.08838834764831845f;

  v8i aq0 = {0,0,0,0,0,0,0,0}, aq1 = {0,0,0,0,0,0,0,0};
  {
    const size_t rowg = ((size_t)bh * N_ + (size_t)qt * QT_ + wave * 16 + ln) * D_;
#pragma unroll
    for (int p = 0; p < 4; ++p) {
      int2 a = *(const int2*)(q8 + rowg + (p >> 1) * 32 + hi * 16 + (p & 1) * 8);
      int2 b = *(const int2*)(q8 + rowg + 64 + (p >> 1) * 32 + hi * 16 + (p & 1) * 8);
      aq0[2 * p] = a.x; aq0[2 * p + 1] = a.y;
      aq1[2 * p] = b.x; aq1[2 * p + 1] = b.y;
    }
  }
  const float qs = qscale[bh * 32 + qt * 2 + (wave >> 2)];
  const float ksv = kscale[bh * 16 + ln];

  float m_run[8], l_run[8], acc[8][8];
#pragma unroll
  for (int r = 0; r < 8; ++r) { m_run[r] = -1e30f; l_run[r] = 0.f; }
#pragma unroll
  for (int dd = 0; dd < 8; ++dd)
#pragma unroll
    for (int r = 0; r < 8; ++r) acc[dd][r] = 0.f;

  stage_tile_async(k8, v8t, vscale, smem, bh, 0, 0, tid);

  for (int kt = 0; kt < NT_; ++kt) {
    const int cur = kt & 1;
    if (kt + 1 < NT_) {
      stage_tile_async(k8, v8t, vscale, smem, bh, kt + 1, cur ^ 1, tid);
      wait_async_le9();
    } else {
      wait_async_le0();
    }
    __syncthreads();

    const char* kbuf = smem + SM_K + cur * 16384;
    const char* vbuf = smem + SM_VT + cur * 16384;
    const float* vsb = (const float*)(smem + SM_VS + cur * 512);
    char* pbuf = smem + SM_P + wave * 2048;

    const float kst =
        __int_as_float(__builtin_amdgcn_readlane(__float_as_int(ksv), kt));
    const float sc = qs * kst * sm_scale;

    float ptl[8][8];
    v8i ci_prev;
#pragma unroll
    for (int s = 0; s < 8; ++s) {
      const int keyl = s * 16 + ln;
      int4 b0 = *(const int4*)(kbuf + keyl * D_ + hi * 16);
      int4 b1 = *(const int4*)(kbuf + keyl * D_ + 32 + hi * 16);
      int4 b2 = *(const int4*)(kbuf + keyl * D_ + 64 + hi * 16);
      int4 b3 = *(const int4*)(kbuf + keyl * D_ + 96 + hi * 16);
      v8i bk0 = {b0.x, b0.y, b0.z, b0.w, b1.x, b1.y, b1.z, b1.w};
      v8i bk1 = {b2.x, b2.y, b2.z, b2.w, b3.x, b3.y, b3.z, b3.w};
      v8i ci = {0, 0, 0, 0, 0, 0, 0, 0};
      ci = __builtin_amdgcn_wmma_i32_16x16x64_iu8(true, aq0, true, bk0, ci, false, false);
      ci = __builtin_amdgcn_wmma_i32_16x16x64_iu8(true, aq1, true, bk1, ci, false, false);
      if (s > 0) {
#pragma unroll
        for (int r = 0; r < 8; ++r) ptl[s - 1][r] = (float)ci_prev[r] * sc;
      }
      ci_prev = ci;
    }
#pragma unroll
    for (int r = 0; r < 8; ++r) ptl[7][r] = (float)ci_prev[r] * sc;

    float mnew[8], corr[8];
#pragma unroll
    for (int r = 0; r < 8; ++r) {
      float t = ptl[0][r];
#pragma unroll
      for (int s = 1; s < 8; ++s) t = fmaxf(t, ptl[s][r]);
      t = hmax16(t);
      mnew[r] = fmaxf(m_run[r], t);
      corr[r] = __expf(m_run[r] - mnew[r]);
      m_run[r] = mnew[r];
    }
#pragma unroll
    for (int dd = 0; dd < 8; ++dd)
#pragma unroll
      for (int r = 0; r < 8; ++r) acc[dd][r] *= corr[r];

    float rs[8];
#pragma unroll
    for (int r = 0; r < 8; ++r) {
      float sump = 0.f, rmax = 0.f;
#pragma unroll
      for (int s = 0; s < 8; ++s) {
        const float p = __expf(ptl[s][r] - mnew[r]);
        sump += p;
        const float pv = p * vsb[s * 16 + ln];
        ptl[s][r] = pv;
        rmax = fmaxf(rmax, pv);
      }
      l_run[r] = l_run[r] * corr[r] + hsum16(sump);
      rs[r] = hmax16(rmax);
    }

#pragma unroll
    for (int r = 0; r < 8; ++r) {
      const float inv = rs[r] > 0.f ? 448.f / rs[r] : 0.f;
#pragma unroll
      for (int s = 0; s < 8; ++s)
        pbuf[(r + hi * 8) * 128 + s * 16 + ln] = (char)f32_to_fp8(ptl[s][r] * inv);
    }

    v16i ap = {0,0,0,0,0,0,0,0,0,0,0,0,0,0,0,0};
#pragma unroll
    for (int c = 0; c < 2; ++c)
#pragma unroll
      for (int p = 0; p < 4; ++p) {
        const int2 w = *(const int2*)(pbuf + ln * 128 + c * 64 + (p >> 1) * 32 + hi * 16 + (p & 1) * 8);
        ap[c * 8 + 2 * p] = w.x;
        ap[c * 8 + 2 * p + 1] = w.y;
      }

#pragma unroll
    for (int dd = 0; dd < 8; ++dd) {
      v16i bv = {0,0,0,0,0,0,0,0,0,0,0,0,0,0,0,0};
#pragma unroll
      for (int g = 0; g < 4; ++g) {
        const int4 w = *(const int4*)(vbuf + (dd * 16 + ln) * KT_ + g * 32 + hi * 16);
        bv[4 * g] = w.x; bv[4 * g + 1] = w.y; bv[4 * g + 2] = w.z; bv[4 * g + 3] = w.w;
      }
      v8f c0 = {0.f, 0.f, 0.f, 0.f, 0.f, 0.f, 0.f, 0.f};
      v8f res = __builtin_amdgcn_wmma_f32_16x16x128_fp8_fp8(ap, bv, (short)0, c0, false, false);
#pragma unroll
      for (int r = 0; r < 8; ++r) acc[dd][r] += (rs[r] * (1.0f / 448.0f)) * res[r];
    }
    __syncthreads();
  }

#pragma unroll
  for (int dd = 0; dd < 8; ++dd)
#pragma unroll
    for (int r = 0; r < 8; ++r) {
      const int m = qt * QT_ + wave * 16 + r + hi * 8;
      out[((size_t)bh * N_ + m) * D_ + dd * 16 + ln] =
          __float2bfloat16(acc[dd][r] / l_run[r]);
    }
}


extern "C" void kernel_launch(void* const* d_in, const int* in_sizes, int n_in,
                              void* d_out, int out_size, void* d_ws, size_t ws_size,
                              hipStream_t stream) {
  (void)in_sizes; (void)n_in; (void)out_size; (void)ws_size;
  const float* q = (const float*)d_in[0];
  const float* k = (const float*)d_in[1];
  const float* v = (const float*)d_in[2];
  __hip_bfloat16* out = (__hip_bfloat16*)d_out;

  char* ws = (char*)d_ws;
  signed char* q8  = (signed char*)ws;
  signed char* k8  = q8 + (size_t)BH_ * N_ * D_;
  signed char* v8t = k8 + (size_t)BH_ * N_ * D_;
  float* qscale = (float*)(v8t + (size_t)BH_ * N_ * D_);
  float* kscale = qscale + BH_ * 32;
  float* vscale = kscale + BH_ * 16;
  float* kmean  = vscale + BH_ * N_;

  kmean_kernel<<<BH_, 128, 0, stream>>>(k, kmean);
  quantk_kernel<<<BH_ * 16, 256, 0, stream>>>(k, kmean, k8, kscale);
  quantq_kernel<<<BH_ * 32, 256, 0, stream>>>(q, q8, qscale);
  quantv_kernel<<<BH_ * N_ / 8, 256, 0, stream>>>(v, v8t, vscale);
  attn_fwd_kernel<<<dim3(N_ / QT_, BH_), 256, SMEM_BYTES, stream>>>(
      q8, k8, v8t, qscale, kscale, vscale, out);
}
